// MHA_60936995995787
// MI455X (gfx1250) — matrix operand order repaired and checked by reading, not compiled, not run on hardware
//
#include <hip/hip_runtime.h>
#include <hip/hip_bf16.h>

// ---------------------------------------------------------------------------
// Causal MHA forward for MI455X (gfx1250, wave32, WMMA).
// fp32->bf16 convert, fused QKV GEMM (double-buffered async-LDS staging),
// flash-attention (async-LDS staging, DPP softmax reductions), out projection.
// ---------------------------------------------------------------------------

typedef __bf16 bf16;
typedef __attribute__((ext_vector_type(16))) __bf16 v16bf;
typedef __attribute__((ext_vector_type(4)))  __bf16 v4bf;
typedef __attribute__((ext_vector_type(8)))  float  v8f;

union Frag {          // one WMMA A/B operand: 16 bf16 = 32 bytes = 2x uint4
  v16bf v;
  uint4 q[2];
};

#define B_  32
#define L_  512
#define D_  4096
#define H_  32
#define HD_ 128
#define M_  (B_ * L_)     // 16384 rows
#define N3_ (3 * D_)      // fused QKV output columns

static __device__ __forceinline__ v8f wmma_bf16(const Frag& a, const Frag& b, v8f c) {
  return __builtin_amdgcn_wmma_f32_16x16x32_bf16(false, a.v, false, b.v,
                                                 (short)0, c, false, false);
}

// ---- CDNA5 async global->LDS copy (ASYNCcnt-tracked, no VGPR round trip) ---
static __device__ __forceinline__ unsigned lds_off32(const void* p) {
  // generic pointers to LDS carry the LDS byte offset in their low 32 bits
  return (unsigned)(uintptr_t)p;
}
static __device__ __forceinline__ void async_copy_b128(void* lds_dst,
                                                       const void* gsrc) {
  asm volatile("global_load_async_to_lds_b128 %0, %1, off"
               :
               : "v"(lds_off32(lds_dst)), "v"((unsigned long long)(uintptr_t)gsrc)
               : "memory");
}
static __device__ __forceinline__ void wait_async0() {
  asm volatile("s_wait_asynccnt 0x0" ::: "memory");
}

// ---- DPP 16-lane all-reduce (keeps softmax off the LDS pipe) ---------------
template <int CTRL>
static __device__ __forceinline__ float dpp_rot(float x) {
  int r = __builtin_amdgcn_update_dpp(__builtin_bit_cast(int, x),
                                      __builtin_bit_cast(int, x),
                                      CTRL, 0xf, 0xf, true);
  return __builtin_bit_cast(float, r);
}
static __device__ __forceinline__ float row16_max(float x) {
  x = fmaxf(x, dpp_rot<0x121>(x));   // row_ror:1
  x = fmaxf(x, dpp_rot<0x122>(x));   // row_ror:2
  x = fmaxf(x, dpp_rot<0x124>(x));   // row_ror:4
  x = fmaxf(x, dpp_rot<0x128>(x));   // row_ror:8
  return x;
}
static __device__ __forceinline__ float row16_sum(float x) {
  x += dpp_rot<0x121>(x);
  x += dpp_rot<0x122>(x);
  x += dpp_rot<0x124>(x);
  x += dpp_rot<0x128>(x);
  return x;
}

static __device__ __forceinline__ float fast_exp2(float x) {
#if __has_builtin(__builtin_amdgcn_exp2f)
  return __builtin_amdgcn_exp2f(x);
#else
  return exp2f(x);
#endif
}

// ------------------------------- convert -----------------------------------
__global__ void cvt_f32_bf16_kernel(const float* __restrict__ src,
                                    bf16* __restrict__ dst, int n4) {
  for (int i = blockIdx.x * blockDim.x + threadIdx.x; i < n4;
       i += gridDim.x * blockDim.x) {
    float4 f = ((const float4*)src)[i];
    v4bf o;
    o[0] = (bf16)f.x; o[1] = (bf16)f.y; o[2] = (bf16)f.z; o[3] = (bf16)f.w;
    ((v4bf*)dst)[i] = o;
  }
}

// ------------------------------- GEMM --------------------------------------
// C[M,N] = A[M,K] * W[N,K]^T (+bias).  A,W bf16 row-major (K contiguous).
// Block tile 128x128x32, double-buffered async-LDS staging, 8 waves (2Mx4N),
// 64x32 output per wave -> 8 WMMA accumulators.
// EPI==0: fused QKV epilogue -> Q,K [B,H,L,HD] bf16, V transposed [B,H,HD,L].
// EPI==1: f32 epilogue -> Fout[M,N] + bias.
template <int EPI>
__global__ __launch_bounds__(256) void gemm_kernel(
    const bf16* __restrict__ A, const bf16* __restrict__ W,
    const float* __restrict__ b0, const float* __restrict__ b1,
    const float* __restrict__ b2, bf16* __restrict__ Qb, bf16* __restrict__ Kb,
    bf16* __restrict__ Vtb, float* __restrict__ Fout, int M, int N, int K) {
  constexpr int BM = 128, BN = 128, BK = 32;
  __shared__ __align__(16) bf16 As[2][BM * BK];   // 2 x 8 KB
  __shared__ __align__(16) bf16 Bs[2][BN * BK];   // 2 x 8 KB

  const int tid  = threadIdx.x;
  const int wave = tid >> 5;
  const int lane = tid & 31;
  const int hi8  = (lane >> 4) & 1;   // upper half-wave
  const int l16  = lane & 15;
  const int wm   = wave & 1;          // 0..1 (M dir)
  const int wn   = wave >> 1;         // 0..3 (N dir)
  const int m0   = blockIdx.y * BM;
  const int n0   = blockIdx.x * BN;

  v8f acc[4][2];
#pragma unroll
  for (int i = 0; i < 4; ++i)
#pragma unroll
    for (int j = 0; j < 2; ++j)
#pragma unroll
      for (int e = 0; e < 8; ++e) acc[i][j][e] = 0.0f;

  auto stage = [&](int buf, int kt) {
    const bf16* Ag = A + (size_t)m0 * K + kt * BK;
    const bf16* Wg = W + (size_t)n0 * K + kt * BK;
#pragma unroll
    for (int i = 0; i < 2; ++i) {            // 512 16B chunks per tile
      int idx = tid + i * 256;
      int r = idx >> 2, c = (idx & 3) << 3;  // 4 chunks per 32-half row
      async_copy_b128(&As[buf][r * BK + c], Ag + (size_t)r * K + c);
      async_copy_b128(&Bs[buf][r * BK + c], Wg + (size_t)r * K + c);
    }
  };

  const int nkt = K / BK;
  stage(0, 0);
  for (int kt = 0; kt < nkt; ++kt) {
    const int buf = kt & 1;
    wait_async0();        // this wave's pending stage is done
    __syncthreads();      // everyone's stage done; prev tile fully consumed
    if (kt + 1 < nkt) stage(buf ^ 1, kt + 1);

    Frag af[4], bfr[2];
#pragma unroll
    for (int mt = 0; mt < 4; ++mt) {         // A: lanes<16 K{0..7,16..23}
      int row = wm * 64 + mt * 16 + l16;
      int kb  = hi8 * 8;
      af[mt].q[0] = *(const uint4*)&As[buf][row * BK + kb];
      af[mt].q[1] = *(const uint4*)&As[buf][row * BK + kb + 16];
    }
#pragma unroll
    for (int nt = 0; nt < 2; ++nt) {         // B: lanes<16 K{0..7,16..23}
      int col = wn * 32 + nt * 16 + l16;
      int kb  = hi8 * 8;
      bfr[nt].q[0] = *(const uint4*)&Bs[buf][col * BK + kb];
      bfr[nt].q[1] = *(const uint4*)&Bs[buf][col * BK + kb + 16];
    }
#pragma unroll
    for (int mt = 0; mt < 4; ++mt)
#pragma unroll
      for (int nt = 0; nt < 2; ++nt)
        acc[mt][nt] = wmma_bf16(af[mt], bfr[nt], acc[mt][nt]);
  }

  // epilogue: element (row = i + hi8*8 within tile, col = l16)
#pragma unroll
  for (int mt = 0; mt < 4; ++mt) {
#pragma unroll
    for (int nt = 0; nt < 2; ++nt) {
      int n_g = n0 + wn * 32 + nt * 16 + l16;
      if (EPI == 0) {
        int which = n_g >> 12;          // 0=Q 1=K 2=V
        int d = n_g & (D_ - 1);
        int h = d >> 7, hd = d & (HD_ - 1);
        const float* bias = (which == 0) ? b0 : (which == 1) ? b1 : b2;
        float bb = bias[d];
        bf16* dst = (which == 0) ? Qb : (which == 1) ? Kb : Vtb;
#pragma unroll
        for (int i = 0; i < 8; ++i) {
          int m_g = m0 + wm * 64 + mt * 16 + i + hi8 * 8;
          int bt = m_g >> 9, lr = m_g & (L_ - 1);
          float v = acc[mt][nt][i] + bb;
          if (which < 2)
            dst[(((size_t)bt * H_ + h) * L_ + lr) * HD_ + hd] = (bf16)v;
          else
            dst[(((size_t)bt * H_ + h) * HD_ + hd) * L_ + lr] = (bf16)v;
        }
      } else {
        float bb = b0[n_g];
#pragma unroll
        for (int i = 0; i < 8; ++i) {
          int m_g = m0 + wm * 64 + mt * 16 + i + hi8 * 8;
          Fout[(size_t)m_g * N + n_g] = acc[mt][nt][i] + bb;
        }
      }
    }
  }
}

// ----------------------------- attention -----------------------------------
// grid: (L/128, H, B); 8 waves; wave w owns q rows [q0, q0+16).
// k streamed in 64-wide blocks via async-LDS; online softmax in exp2 domain.
__global__ __launch_bounds__(256) void attn_kernel(
    const bf16* __restrict__ Qb, const bf16* __restrict__ Kb,
    const bf16* __restrict__ Vtb, bf16* __restrict__ Ob) {
  __shared__ __align__(16) bf16 Ks[64 * 128];      // [kpos][hd]  16 KB
  __shared__ __align__(16) bf16 Vs[128 * 64];      // [hd][kpos]  16 KB
  __shared__ __align__(16) bf16 Ps[8][16 * 64];    // per-wave P  16 KB

  const int tid  = threadIdx.x;
  const int wave = tid >> 5;
  const int lane = tid & 31;
  const int hi8  = (lane >> 4) & 1;
  const int l16  = lane & 15;
  const int qblock = blockIdx.x, h = blockIdx.y, b = blockIdx.z;
  const int q0 = qblock * 128 + wave * 16;

  const size_t head = (size_t)b * H_ + h;
  const bf16* Qh = Qb + head * L_ * HD_;
  const bf16* Kh = Kb + head * L_ * HD_;
  const bf16* Vh = Vtb + head * HD_ * L_;

  // Q fragments (16 rows x full HD=128), kept in registers
  Frag qf[4];
  {
    const bf16* qrow = Qh + (size_t)(q0 + l16) * HD_;
#pragma unroll
    for (int ks = 0; ks < 4; ++ks) {
      int kb = ks * 32 + hi8 * 8;
      qf[ks].q[0] = *(const uint4*)(qrow + kb);
      qf[ks].q[1] = *(const uint4*)(qrow + kb + 16);
    }
  }

  v8f Oacc[8];
  float mrow[8], lrow[8];
#pragma unroll
  for (int t = 0; t < 8; ++t) {
#pragma unroll
    for (int e = 0; e < 8; ++e) Oacc[t][e] = 0.0f;
    mrow[t] = -1e30f;
    lrow[t] = 0.0f;
  }

  const float SCALE = 0.08838834764831845f * 1.44269504088896340f; // log2e/sqrt(128)

  const int nkb = 2 * qblock + 2;            // causal: only to the diagonal
  for (int kb = 0; kb < nkb; ++kb) {
    // cooperative async staging (all waves; barriers wave-uniform)
    const bf16* Kg = Kh + (size_t)kb * 64 * HD_;
#pragma unroll
    for (int i = 0; i < 4; ++i) {            // K: 64x128 halves
      int idx = tid + i * 256;
      int r = idx >> 4, c = (idx & 15) << 3;
      async_copy_b128(&Ks[r * 128 + c], Kg + (size_t)r * HD_ + c);
    }
    const bf16* Vg = Vh + kb * 64;
#pragma unroll
    for (int i = 0; i < 4; ++i) {            // V^T: 128x64 halves
      int idx = tid + i * 256;
      int r = idx >> 3, c = (idx & 7) << 3;
      async_copy_b128(&Vs[r * 64 + c], Vg + (size_t)r * L_ + c);
    }
    wait_async0();
    __syncthreads();

    if (kb * 64 <= q0 + 15) {                // this wave still below diagonal
      // ---- S = Q * K^T  (16x64, f32) ----
      v8f S[4];
#pragma unroll
      for (int nt = 0; nt < 4; ++nt)
#pragma unroll
        for (int e = 0; e < 8; ++e) S[nt][e] = 0.0f;
#pragma unroll
      for (int nt = 0; nt < 4; ++nt) {
#pragma unroll
        for (int ks = 0; ks < 4; ++ks) {
          Frag kf;
          int col = nt * 16 + l16;
          int kc  = ks * 32 + hi8 * 8;
          kf.q[0] = *(const uint4*)&Ks[col * 128 + kc];
          kf.q[1] = *(const uint4*)&Ks[col * 128 + kc + 16];
          S[nt] = wmma_bf16(qf[ks], kf, S[nt]);
        }
      }
      // ---- causal mask on diagonal-crossing blocks ----
      if (kb * 64 + 63 > q0) {
#pragma unroll
        for (int nt = 0; nt < 4; ++nt) {
          int kpos = kb * 64 + nt * 16 + l16;
#pragma unroll
          for (int i = 0; i < 8; ++i) {
            int qpos = q0 + i + hi8 * 8;
            if (kpos > qpos) S[nt][i] = -1e30f;
          }
        }
      }
      // ---- online softmax (row = VGPR idx within half-wave; DPP reduce) ----
      float alpha[8], rs[8];
#pragma unroll
      for (int i = 0; i < 8; ++i) {
        float ml = fmaxf(fmaxf(S[0][i], S[1][i]), fmaxf(S[2][i], S[3][i]));
        ml = row16_max(ml);
        float mn = fmaxf(mrow[i], ml);
        alpha[i] = fast_exp2((mrow[i] - mn) * SCALE);
        mrow[i]  = mn;
        rs[i]    = 0.0f;
      }
#pragma unroll
      for (int nt = 0; nt < 4; ++nt) {
#pragma unroll
        for (int i = 0; i < 8; ++i) {
          float p = fast_exp2((S[nt][i] - mrow[i]) * SCALE);
          rs[i] += p;
          Ps[wave][(i + hi8 * 8) * 64 + nt * 16 + l16] = (bf16)p;
        }
      }
#pragma unroll
      for (int i = 0; i < 8; ++i) {
        rs[i] = row16_sum(rs[i]);
        lrow[i] = lrow[i] * alpha[i] + rs[i];
      }
#pragma unroll
      for (int t = 0; t < 8; ++t)
#pragma unroll
        for (int i = 0; i < 8; ++i) Oacc[t][i] *= alpha[i];

      // ---- O += P * V  (contraction over 64 k-positions) ----
      Frag pf[2];
#pragma unroll
      for (int ks2 = 0; ks2 < 2; ++ks2) {
        int kc = ks2 * 32 + hi8 * 8;
        pf[ks2].q[0] = *(const uint4*)&Ps[wave][l16 * 64 + kc];
        pf[ks2].q[1] = *(const uint4*)&Ps[wave][l16 * 64 + kc + 16];
      }
#pragma unroll
      for (int t = 0; t < 8; ++t) {
#pragma unroll
        for (int ks2 = 0; ks2 < 2; ++ks2) {
          Frag vf;
          int col = t * 16 + l16;
          int kc2 = ks2 * 32 + hi8 * 8;
          vf.q[0] = *(const uint4*)&Vs[col * 64 + kc2];
          vf.q[1] = *(const uint4*)&Vs[col * 64 + kc2 + 16];
          Oacc[t] = wmma_bf16(pf[ks2], vf, Oacc[t]);
        }
      }
    }
    __syncthreads();   // protect Ks/Vs from next iteration's staging
  }

  // ---- write O (bf16, [B,L,D] layout, head h occupies cols h*128..) ----
#pragma unroll
  for (int t = 0; t < 8; ++t) {
    int col = h * HD_ + t * 16 + l16;
#pragma unroll
    for (int i = 0; i < 8; ++i) {
      int q = q0 + i + hi8 * 8;
      float v = Oacc[t][i] / lrow[i];
      Ob[((size_t)b * L_ + q) * D_ + col] = (bf16)v;
    }
  }
}

// ----------------------------- launch --------------------------------------
extern "C" void kernel_launch(void* const* d_in, const int* in_sizes, int n_in,
                              void* d_out, int out_size, void* d_ws,
                              size_t ws_size, hipStream_t stream) {
  const float* x  = (const float*)d_in[0];
  const float* Wq = (const float*)d_in[1];
  const float* bq = (const float*)d_in[2];
  const float* Wk = (const float*)d_in[3];
  const float* bk = (const float*)d_in[4];
  const float* Wv = (const float*)d_in[5];
  const float* bv = (const float*)d_in[6];
  const float* Wo = (const float*)d_in[7];
  const float* bo = (const float*)d_in[8];
  float* out = (float*)d_out;

  char* ws = (char*)d_ws;
  size_t off = 0;
  bf16* xb  = (bf16*)(ws + off); off += (size_t)M_ * D_ * 2;       // 128 MB
  bf16* Wb  = (bf16*)(ws + off); off += (size_t)3 * D_ * D_ * 2;   //  96 MB
  bf16* Wob = (bf16*)(ws + off); off += (size_t)D_ * D_ * 2;       //  32 MB
  bf16* Qb  = (bf16*)(ws + off); off += (size_t)M_ * D_ * 2;       // 128 MB
  bf16* Kb  = (bf16*)(ws + off); off += (size_t)M_ * D_ * 2;       // 128 MB
  bf16* Vtb = (bf16*)(ws + off); off += (size_t)M_ * D_ * 2;       // 128 MB
  bf16* Ob  = (bf16*)(ws + off); off += (size_t)M_ * D_ * 2;       // 128 MB

  auto cvt = [&](const float* s, bf16* d, size_t n) {
    int n4 = (int)(n / 4);
    int blocks = (n4 + 255) / 256;
    if (blocks > 16384) blocks = 16384;
    cvt_f32_bf16_kernel<<<blocks, 256, 0, stream>>>(s, d, n4);
  };
  cvt(x,  xb,  (size_t)M_ * D_);
  cvt(Wq, Wb,                   (size_t)D_ * D_);
  cvt(Wk, Wb + (size_t)D_ * D_, (size_t)D_ * D_);
  cvt(Wv, Wb + (size_t)2 * D_ * D_, (size_t)D_ * D_);
  cvt(Wo, Wob, (size_t)D_ * D_);

  // fused QKV projection: [16384,4096] x [12288,4096]^T
  gemm_kernel<0><<<dim3(N3_ / 128, M_ / 128), 256, 0, stream>>>(
      xb, Wb, bq, bk, bv, Qb, Kb, Vtb, nullptr, M_, N3_, D_);

  // flash attention, causal
  attn_kernel<<<dim3(L_ / 128, H_, B_), 256, 0, stream>>>(Qb, Kb, Vtb, Ob);

  // output projection: [16384,4096] x [4096,4096]^T -> f32
  gemm_kernel<1><<<dim3(D_ / 128, M_ / 128), 256, 0, stream>>>(
      Ob, Wob, bo, nullptr, nullptr, nullptr, nullptr, nullptr, out, M_, D_, D_);
}
